// OptimizedTransformerEncoder_43619687858422
// MI455X (gfx1250) — compile-verified, matrix operand order checked by reading, not run on hardware
//
#include <hip/hip_runtime.h>
#include <hip/hip_bf16.h>
#include <stdint.h>

// ---------------------------------------------------------------------------
// Deformable-DETR encoder, 6 layers, B=8, N=10000 (100x100), D=256, FF=1024.
// All GEMMs: bf16 WMMA (v_wmma_f32_16x16x32_bf16), f32 accumulate, with
// GLOBAL_LOAD_ASYNC_TO_LDS_B128 double-buffered tile staging (ASYNCcnt).
// Per layer:
//   1) proj = x @ [Wv|Woff|Wa] (+bias)   -> WMMA GEMM, N=352 (weights padded to 384)
//   2) deformable bilinear sampling      -> gather kernel, wave per (token, head)
//   3) attn = sampled @ Wo (+bo)         -> WMMA GEMM, N=256
//   4) x = LN(x + attn)                  -> wave-per-row LN (wave32 shfl reduce)
//   5) h = gelu(x @ W1 + b1) (bf16 out)  -> WMMA GEMM, N=1024, K=256
//   6) f = h @ W2 + b2                   -> WMMA GEMM, N=256,  K=1024
//   7) x = LN(x + f)
// ---------------------------------------------------------------------------

typedef __attribute__((ext_vector_type(16))) __bf16 v16bf;
typedef __attribute__((ext_vector_type(8)))  float  v8f;

#define D_MODEL 256
#define FF_DIM  1024
#define HEADS   8
#define PTS     4
#define HD      32
#define GRID_W  100
#define GRID_H  100
#define TOK     10000     // per-batch tokens
#define NPROJ   352       // 256 (v) + 64 (off) + 32 (attn)
#define NPROJ_P 384       // padded to a multiple of TN (zero rows 352..383)

#define TM 128
#define TN 64
#define TK 64
#define LDS_S 72          // padded row stride (uint16) -> conflict-free b128 reads

__device__ __forceinline__ uint16_t f2bf(float f) {
    uint32_t u = __float_as_uint(f);
    u += 0x7FFFu + ((u >> 16) & 1u);     // round-to-nearest-even
    return (uint16_t)(u >> 16);
}

union Frag { v16bf v; uint4 q[2]; };

// ---- CDNA5 async global->LDS copy (16B per lane, tracked by ASYNCcnt) ----
__device__ __forceinline__ void async_copy16(const uint16_t* g, uint16_t* l) {
    uint32_t lds  = (uint32_t)(uintptr_t)l;                 // low 32 bits = LDS offset
    unsigned long long ga = (unsigned long long)(uintptr_t)g;
    asm volatile("global_load_async_to_lds_b128 %0, %1, off"
                 :: "v"(lds), "v"(ga) : "memory");
}
__device__ __forceinline__ void wait_async0() {
    asm volatile("s_wait_asynccnt 0x0" ::: "memory");
}

// ---------------------------------------------------------------------------
// bf16 WMMA GEMM:  C[M,N] = A[M,K] * Bt[N,K]^T + bias
// Requirements: M % 128 == 0, N % 64 == 0 for the *staged* B buffer (weights
// padded), K % 64 == 0. Epilogue guards stores against logical N.
// EPI: 0 = f32 store, 2 = exact GELU then bf16 store
// ---------------------------------------------------------------------------
template <int EPI>
__global__ __launch_bounds__(256)
void gemm_bf16_wmma(const uint16_t* __restrict__ A,
                    const uint16_t* __restrict__ Bt,
                    const float*    __restrict__ bias,
                    void*           __restrict__ Cptr,
                    int M, int N, int K, int ldc)
{
    __shared__ __align__(16) uint16_t lA[2][TM * LDS_S];
    __shared__ __align__(16) uint16_t lB[2][TN * LDS_S];

    const int tid   = threadIdx.x;
    const int wave  = tid >> 5;
    const int lane  = tid & 31;
    const int mBase = blockIdx.y * TM;
    const int nBase = blockIdx.x * TN;
    const int wRow  = (wave >> 1) * 32;   // 0,32,64,96
    const int wCol  = (wave & 1)  * 32;   // 0,32
    const int lh    = lane >> 4;          // K-half select (ISA 16-bit A layout)
    const int l15   = lane & 15;

    v8f acc[2][2] = {};

    // stage one 128x64 A tile + 64x64 B tile into LDS buffer `buf`
    auto stage = [&](int buf, int k0) {
        #pragma unroll
        for (int i = 0; i < 4; ++i) {                 // A: 1024 chunks of 16B
            int c = tid + 256 * i;
            int row = c >> 3;                         // 0..127
            int col = (c & 7) * 8;                    // 0..56
            async_copy16(A + (size_t)(mBase + row) * K + k0 + col,
                         &lA[buf][row * LDS_S + col]);
        }
        #pragma unroll
        for (int i = 0; i < 2; ++i) {                 // B: 512 chunks of 16B
            int c = tid + 256 * i;
            int row = c >> 3;                         // 0..63
            int col = (c & 7) * 8;
            async_copy16(Bt + (size_t)(nBase + row) * K + k0 + col,
                         &lB[buf][row * LDS_S + col]);
        }
    };

    stage(0, 0);
    const int nt = K / TK;
    for (int kt = 0; kt < nt; ++kt) {
        wait_async0();                 // this wave's part of tile kt landed in LDS
        __syncthreads();               // whole tile kt visible; prev reads retired
        if (kt + 1 < nt) stage((kt + 1) & 1, (kt + 1) * TK);   // overlap copy
        const int buf = kt & 1;

        #pragma unroll
        for (int kk = 0; kk < 2; ++kk) {              // two 16x16x32 K-slices
            Frag a[2], b[2];
            #pragma unroll
            for (int i = 0; i < 2; ++i) {
                const uint16_t* p = &lA[buf][(wRow + 16 * i + l15) * LDS_S + kk * 32 + lh * 8];
                a[i].q[0] = *(const uint4*)p;
                a[i].q[1] = *(const uint4*)(p + 16);
            }
            #pragma unroll
            for (int j = 0; j < 2; ++j) {
                const uint16_t* p = &lB[buf][(wCol + 16 * j + l15) * LDS_S + kk * 32 + lh * 8];
                b[j].q[0] = *(const uint4*)p;
                b[j].q[1] = *(const uint4*)(p + 16);
            }
            #pragma unroll
            for (int i = 0; i < 2; ++i)
                #pragma unroll
                for (int j = 0; j < 2; ++j)
                    acc[i][j] = __builtin_amdgcn_wmma_f32_16x16x32_bf16(
                        false, a[i].v, false, b[j].v, (short)0, acc[i][j], false, false);
        }
        __syncthreads();               // all reads of buf done before its rewrite
    }

    // ---- epilogue: VGPR e holds row (lh*8+e), col l15 ----
    #pragma unroll
    for (int i = 0; i < 2; ++i) {
        #pragma unroll
        for (int j = 0; j < 2; ++j) {
            int colg = nBase + wCol + 16 * j + l15;
            if (colg >= N) continue;
            float bval = bias[colg];
            #pragma unroll
            for (int e = 0; e < 8; ++e) {
                int rowg = mBase + wRow + 16 * i + lh * 8 + e;
                float v = acc[i][j][e] + bval;
                if (EPI == 2) v = 0.5f * v * (1.0f + erff(v * 0.70710678118654752f));
                if (EPI == 0)
                    ((float*)Cptr)[(size_t)rowg * ldc + colg] = v;
                else
                    ((uint16_t*)Cptr)[(size_t)rowg * ldc + colg] = f2bf(v);
            }
        }
    }
}

// ---------------------------------------------------------------------------
// x = src + 2D sine positional encoding; emit f32 and bf16 copies
// ---------------------------------------------------------------------------
__global__ __launch_bounds__(256)
void pos_add(const float* __restrict__ src, float* __restrict__ xf,
             uint16_t* __restrict__ xb, int total)
{
    int id = blockIdx.x * 256 + threadIdx.x;
    if (id >= total) return;
    int m = id >> 8;
    int d = id & 255;
    int n = m % TOK;
    int yy = n / GRID_W, xx = n % GRID_W;
    int pos, dd;
    if (d < 128) { pos = yy; dd = d; } else { pos = xx; dd = d - 128; }
    int i = dd >> 1;
    float freq = expf(-9.210340371976184f * (float)(2 * i) * (1.0f / 128.0f));
    float ang  = (float)pos * freq;
    float pe   = (dd & 1) ? cosf(ang) : sinf(ang);
    float v    = src[id] + pe;
    xf[id] = v;
    xb[id] = f2bf(v);
}

// ---------------------------------------------------------------------------
// Deformable sampling: one wave per (token, head); lane = head channel.
// proj row = [ v(256) | off(64) | attn-logits(32) ], stride NPROJ.
// ---------------------------------------------------------------------------
__global__ __launch_bounds__(256)
void deform_sample(const float* __restrict__ proj, uint16_t* __restrict__ sampled, int M)
{
    int m = blockIdx.x;
    if (m >= M) return;
    int h = threadIdx.x >> 5;
    int c = threadIdx.x & 31;
    const float* row = proj + (size_t)m * NPROJ;

    float lg[4];
    #pragma unroll
    for (int p = 0; p < 4; ++p) lg[p] = row[320 + h * 4 + p];
    float mx = fmaxf(fmaxf(lg[0], lg[1]), fmaxf(lg[2], lg[3]));
    float e[4], s = 0.f;
    #pragma unroll
    for (int p = 0; p < 4; ++p) { e[p] = expf(lg[p] - mx); s += e[p]; }
    float inv = 1.0f / s;

    int b  = m / TOK;
    int n  = m % TOK;
    int yy = n / GRID_W, xx = n % GRID_W;
    const float* vb = proj + (size_t)b * TOK * NPROJ + h * HD + c;

    float acc = 0.f;
    #pragma unroll
    for (int p = 0; p < 4; ++p) {
        float dx = row[256 + (h * 4 + p) * 2 + 0];
        float dy = row[256 + (h * 4 + p) * 2 + 1];
        // px = loc_x*W - 0.5 collapses to xx + dx for this grid
        float px = (float)xx + dx;
        float py = (float)yy + dy;
        float x0 = floorf(px), y0 = floorf(py);
        float wx = px - x0,    wy = py - y0;
        int x0i = min(max((int)x0, 0), GRID_W - 1);
        int x1i = min(x0i + 1, GRID_W - 1);
        int y0i = min(max((int)y0, 0), GRID_H - 1);
        int y1i = min(y0i + 1, GRID_H - 1);
        float v00 = vb[(size_t)(y0i * GRID_W + x0i) * NPROJ];
        float v01 = vb[(size_t)(y0i * GRID_W + x1i) * NPROJ];
        float v10 = vb[(size_t)(y1i * GRID_W + x0i) * NPROJ];
        float v11 = vb[(size_t)(y1i * GRID_W + x1i) * NPROJ];
        float bil = v00 * (1.f - wx) * (1.f - wy) + v01 * wx * (1.f - wy)
                  + v10 * (1.f - wx) * wy        + v11 * wx * wy;
        acc += e[p] * inv * bil;
    }
    sampled[(size_t)m * D_MODEL + h * HD + c] = f2bf(acc);
}

// ---------------------------------------------------------------------------
// x = LayerNorm(xin + delta) ; wave32 per row, 8 channels per lane
// ---------------------------------------------------------------------------
__global__ __launch_bounds__(256)
void add_ln(const float* __restrict__ xin, const float* __restrict__ delta,
            const float* __restrict__ g, const float* __restrict__ be,
            float* __restrict__ xout, uint16_t* __restrict__ xb, int M)
{
    int row  = blockIdx.x * 8 + (threadIdx.x >> 5);
    int lane = threadIdx.x & 31;
    if (row >= M) return;
    const float* xi = xin   + (size_t)row * D_MODEL;
    const float* dl = delta + (size_t)row * D_MODEL;

    float v[8];
    float s = 0.f;
    #pragma unroll
    for (int t = 0; t < 8; ++t) {
        int c = lane + 32 * t;
        v[t] = xi[c] + dl[c];
        s += v[t];
    }
    #pragma unroll
    for (int o = 16; o >= 1; o >>= 1) s += __shfl_xor(s, o, 32);
    float mean = s * (1.0f / 256.0f);
    float q = 0.f;
    #pragma unroll
    for (int t = 0; t < 8; ++t) { float d = v[t] - mean; q += d * d; }
    #pragma unroll
    for (int o = 16; o >= 1; o >>= 1) q += __shfl_xor(q, o, 32);
    float rstd = rsqrtf(q * (1.0f / 256.0f) + 1e-5f);
    #pragma unroll
    for (int t = 0; t < 8; ++t) {
        int c = lane + 32 * t;
        float o = (v[t] - mean) * rstd * g[c] + be[c];
        xout[(size_t)row * D_MODEL + c] = o;
        xb  [(size_t)row * D_MODEL + c] = f2bf(o);
    }
}

// ---------------------------------------------------------------------------
// Weight prep: transpose + bf16 convert; fused [Wv|Woff|Wa] block (zero-padded)
// ---------------------------------------------------------------------------
__global__ __launch_bounds__(256)
void conv_transpose_bf16(const float* __restrict__ src, uint16_t* __restrict__ dst,
                         int K, int Nc)
{
    int id = blockIdx.x * 256 + threadIdx.x;
    if (id >= K * Nc) return;
    int n = id / K, k = id - n * K;
    dst[(size_t)n * K + k] = f2bf(src[(size_t)k * Nc + n]);
}

__global__ __launch_bounds__(256)
void build_wcat(const float* __restrict__ Wv, const float* __restrict__ Woff,
                const float* __restrict__ Wa, uint16_t* __restrict__ dst)
{
    int id = blockIdx.x * 256 + threadIdx.x;   // over 384*256 (padded)
    if (id >= NPROJ_P * D_MODEL) return;
    int n = id / D_MODEL, k = id - n * D_MODEL;
    float v = 0.0f;
    if      (n < 256) v = Wv  [(size_t)k * 256 + n];
    else if (n < 320) v = Woff[(size_t)k * 64  + (n - 256)];
    else if (n < 352) v = Wa  [(size_t)k * 32  + (n - 320)];
    dst[(size_t)n * D_MODEL + k] = f2bf(v);
}

__global__ __launch_bounds__(256)
void build_bcat(const float* __restrict__ bv, const float* __restrict__ boff,
                const float* __restrict__ ba, float* __restrict__ dst)
{
    int id = blockIdx.x * 256 + threadIdx.x;
    if (id >= NPROJ) return;
    dst[id] = (id < 256) ? bv[id] : (id < 320) ? boff[id - 256] : ba[id - 320];
}

// ---------------------------------------------------------------------------
extern "C" void kernel_launch(void* const* d_in, const int* in_sizes, int n_in,
                              void* d_out, int out_size, void* d_ws, size_t ws_size,
                              hipStream_t stream)
{
    const float* src  = (const float*)d_in[0];
    const float* Wv   = (const float*)d_in[1];
    const float* bv   = (const float*)d_in[2];
    const float* Woff = (const float*)d_in[3];
    const float* boff = (const float*)d_in[4];
    const float* Wa   = (const float*)d_in[5];
    const float* ba   = (const float*)d_in[6];
    const float* Wo   = (const float*)d_in[7];
    const float* bo   = (const float*)d_in[8];
    const float* W1   = (const float*)d_in[9];
    const float* b1   = (const float*)d_in[10];
    const float* W2   = (const float*)d_in[11];
    const float* b2   = (const float*)d_in[12];
    const float* g1   = (const float*)d_in[13];
    const float* be1  = (const float*)d_in[14];
    const float* g2   = (const float*)d_in[15];
    const float* be2  = (const float*)d_in[16];

    const int M = in_sizes[0] / D_MODEL;          // 80000 tokens (multiple of 128)
    const int Lnum = in_sizes[1] / (D_MODEL * D_MODEL);

    // ---- workspace carve-up (256B aligned) ----
    uint8_t* ws = (uint8_t*)d_ws;
    size_t off = 0;
    auto carve = [&](size_t bytes) -> uint8_t* {
        off = (off + 255) & ~(size_t)255;
        uint8_t* p = ws + off;
        off += bytes;
        return p;
    };
    uint16_t* wcatT = (uint16_t*)carve((size_t)Lnum * NPROJ_P * D_MODEL * 2);
    uint16_t* woT   = (uint16_t*)carve((size_t)Lnum * D_MODEL * D_MODEL * 2);
    uint16_t* w1T   = (uint16_t*)carve((size_t)Lnum * D_MODEL * FF_DIM * 2);
    uint16_t* w2T   = (uint16_t*)carve((size_t)Lnum * FF_DIM * D_MODEL * 2);
    float*    bcat  = (float*)   carve((size_t)Lnum * NPROJ * 4);
    float*    xf    = (float*)   carve((size_t)M * D_MODEL * 4);
    uint16_t* xb    = (uint16_t*)carve((size_t)M * D_MODEL * 2);
    float*    proj  = (float*)   carve((size_t)M * NPROJ * 4);
    uint16_t* smp   = (uint16_t*)carve((size_t)M * D_MODEL * 2);
    uint16_t* hb    = (uint16_t*)carve((size_t)M * FF_DIM * 2);
    float*    gout  = (float*)   carve((size_t)M * D_MODEL * 4);

    // ---- weight prep (deterministic each call) ----
    for (int l = 0; l < Lnum; ++l) {
        build_wcat<<<(NPROJ_P * D_MODEL + 255) / 256, 256, 0, stream>>>(
            Wv + (size_t)l * D_MODEL * D_MODEL, Woff + (size_t)l * D_MODEL * 64,
            Wa + (size_t)l * D_MODEL * 32, wcatT + (size_t)l * NPROJ_P * D_MODEL);
        build_bcat<<<2, 256, 0, stream>>>(
            bv + (size_t)l * D_MODEL, boff + (size_t)l * 64, ba + (size_t)l * 32,
            bcat + (size_t)l * NPROJ);
        conv_transpose_bf16<<<(D_MODEL * D_MODEL + 255) / 256, 256, 0, stream>>>(
            Wo + (size_t)l * D_MODEL * D_MODEL, woT + (size_t)l * D_MODEL * D_MODEL,
            D_MODEL, D_MODEL);
        conv_transpose_bf16<<<(D_MODEL * FF_DIM + 255) / 256, 256, 0, stream>>>(
            W1 + (size_t)l * D_MODEL * FF_DIM, w1T + (size_t)l * D_MODEL * FF_DIM,
            D_MODEL, FF_DIM);
        conv_transpose_bf16<<<(FF_DIM * D_MODEL + 255) / 256, 256, 0, stream>>>(
            W2 + (size_t)l * FF_DIM * D_MODEL, w2T + (size_t)l * FF_DIM * D_MODEL,
            FF_DIM, D_MODEL);
    }

    // ---- x = src + pos encoding ----
    pos_add<<<(M * D_MODEL + 255) / 256, 256, 0, stream>>>(src, xf, xb, M * D_MODEL);

    const int mt = M / TM;
    dim3 blk(256);

    for (int l = 0; l < Lnum; ++l) {
        const uint16_t* wc = wcatT + (size_t)l * NPROJ_P * D_MODEL;
        const uint16_t* wo = woT   + (size_t)l * D_MODEL * D_MODEL;
        const uint16_t* w1 = w1T   + (size_t)l * D_MODEL * FF_DIM;
        const uint16_t* w2 = w2T   + (size_t)l * FF_DIM * D_MODEL;

        // 1) fused projection: [v | off | attn-logits]
        gemm_bf16_wmma<0><<<dim3(NPROJ_P / TN, mt), blk, 0, stream>>>(
            xb, wc, bcat + (size_t)l * NPROJ, proj, M, NPROJ, D_MODEL, NPROJ);
        // 2) deformable bilinear sampling + softmax-weighted sum
        deform_sample<<<M, 256, 0, stream>>>(proj, smp, M);
        // 3) output projection
        gemm_bf16_wmma<0><<<dim3(D_MODEL / TN, mt), blk, 0, stream>>>(
            smp, wo, bo + (size_t)l * D_MODEL, gout, M, D_MODEL, D_MODEL, D_MODEL);
        // 4) x = LN(x + attn)
        add_ln<<<(M + 7) / 8, 256, 0, stream>>>(
            xf, gout, g1 + (size_t)l * D_MODEL, be1 + (size_t)l * D_MODEL, xf, xb, M);
        // 5) h = gelu(x @ W1 + b1), bf16 out
        gemm_bf16_wmma<2><<<dim3(FF_DIM / TN, mt), blk, 0, stream>>>(
            xb, w1, b1 + (size_t)l * FF_DIM, hb, M, FF_DIM, D_MODEL, FF_DIM);
        // 6) f = h @ W2 + b2
        gemm_bf16_wmma<0><<<dim3(D_MODEL / TN, mt), blk, 0, stream>>>(
            hb, w2, b2 + (size_t)l * D_MODEL, gout, M, D_MODEL, FF_DIM, D_MODEL);
        // 7) x = LN(x + f); last layer writes straight to d_out
        float* xdst = (l == Lnum - 1) ? (float*)d_out : xf;
        add_ln<<<(M + 7) / 8, 256, 0, stream>>>(
            xf, gout, g2 + (size_t)l * D_MODEL, be2 + (size_t)l * D_MODEL, xdst, xb, M);
    }
}
